// CausalSelfAttention_70471823393128
// MI455X (gfx1250) — matrix operand order repaired and checked by reading, not compiled, not run on hardware
//
#include <hip/hip_runtime.h>

// MI455X / gfx1250 causal self-attention forward.
// - All GEMMs via v_wmma_f32_16x16x32_bf16 (fp32 accumulate)
// - Flash-attention inner loop with K/V tiles streamed by the Tensor Data
//   Mover (tensor_load_to_lds, TENSORcnt, double-buffered)
// - RoPE fused into the QKV GEMM epilogue
//
// Workspace layout (bf16 elements): xb | wqkvT | wprojT | qb | kb | vbT | ob
// total ~197 MB (assumed <= ws_size).

typedef __bf16 bf16;
typedef __attribute__((ext_vector_type(16))) __bf16 v16bf;
typedef __attribute__((ext_vector_type(8)))  __bf16 v8bf;
typedef __attribute__((ext_vector_type(8)))  float  v8f;
typedef __attribute__((ext_vector_type(4)))  unsigned int v4ui;
typedef __attribute__((ext_vector_type(8)))  int v8i;
typedef __attribute__((ext_vector_type(4)))  int v4i;

constexpr int Bb = 4, Tt = 2048, Cc = 2048, Hh = 16, Dd = 128;
constexpr int Mm = Bb * Tt;          // 8192 rows
constexpr int Nqkv = 3 * Cc;         // 6144

__device__ inline v8f vzero8() {
  v8f z;
#pragma unroll
  for (int e = 0; e < 8; ++e) z[e] = 0.0f;
  return z;
}

__device__ inline v16bf ld_frag(const bf16* p0, const bf16* p1) {
  v8bf lo = *(const v8bf*)p0;
  v8bf hi = *(const v8bf*)p1;
  v16bf r;
#pragma unroll
  for (int e = 0; e < 8; ++e) { r[e] = lo[e]; r[e + 8] = hi[e]; }
  return r;
}

__device__ inline v8f wmma_bf16(v16bf a, v16bf b, v8f c) {
  return __builtin_amdgcn_wmma_f32_16x16x32_bf16(
      false, a, false, b, (short)0, c, false, false);
}

// Tensor Data Mover: 2D bf16 tile -> LDS, +4 DWORD pad every 64 DWORDs
// (256B row -> 272B LDS stride == 136 bf16 elems).
__device__ inline void tdm_load_tile_bf16(unsigned lds_off, const bf16* gsrc,
                                          unsigned width, unsigned rows,
                                          unsigned row_stride) {
  unsigned long long ga = (unsigned long long)(const void*)gsrc;
  v4ui g0;
  g0[0] = 1u;                                   // count=1, user descriptor
  g0[1] = lds_off;                              // lds_addr (bytes)
  g0[2] = (unsigned)ga;                         // global_addr[31:0]
  g0[3] = (unsigned)((ga >> 32) & 0x01ffffffu)  // global_addr[56:32]
          | (2u << 30);                         // type=2 (image)
  v8i g1;
  g1[0] = (int)((1u << 16)     // data_size = 2 bytes
                | (1u << 20)   // pad_enable
                | (5u << 22)   // pad_interval: 64 DWORDs
                | (3u << 25)); // pad_amount: 4 DWORDs
  g1[1] = (int)((width & 0xffffu) << 16);                        // dim0 lo16
  g1[2] = (int)((width >> 16) | ((rows & 0xffffu) << 16));       // dim0 hi/dim1 lo
  g1[3] = (int)((rows >> 16) | ((width & 0xffffu) << 16));       // dim1 hi/tile0
  g1[4] = (int)(rows & 0xffffu);                                 // tile1, tile2=0
  g1[5] = (int)row_stride;                                       // dim0_stride lo32
  g1[6] = 0;
  g1[7] = 0;
  v4i z4;
#pragma unroll
  for (int e = 0; e < 4; ++e) z4[e] = 0;
#if defined(__clang_major__) && (__clang_major__ >= 23)
  v8i z8;
#pragma unroll
  for (int e = 0; e < 8; ++e) z8[e] = 0;
  __builtin_amdgcn_tensor_load_to_lds(g0, g1, z4, z4, z8, 0);
#else
  __builtin_amdgcn_tensor_load_to_lds(g0, g1, z4, z4, 0);
#endif
}

// ---------------------------------------------------------------- converts
__global__ __launch_bounds__(256) void cast_bf16_kernel(
    const float* __restrict__ in, bf16* __restrict__ out, int n) {
  for (int i = blockIdx.x * 256 + threadIdx.x; i < n; i += gridDim.x * 256)
    out[i] = (bf16)in[i];
}

// in[R][N] fp32 -> out[N][R] bf16 (tiled transpose through LDS)
__global__ __launch_bounds__(256) void transpose_cast_kernel(
    const float* __restrict__ in, bf16* __restrict__ out, int R, int N) {
  __shared__ float tile[32][33];
  int n0 = blockIdx.x * 32, r0 = blockIdx.y * 32;
#pragma unroll
  for (int yy = threadIdx.y; yy < 32; yy += 8)
    tile[yy][threadIdx.x] = in[(size_t)(r0 + yy) * N + n0 + threadIdx.x];
  __syncthreads();
#pragma unroll
  for (int yy = threadIdx.y; yy < 32; yy += 8)
    out[(size_t)(n0 + yy) * R + r0 + threadIdx.x] = (bf16)tile[threadIdx.x][yy];
}

// ---------------------------------------------------------------- QKV GEMM + RoPE
#define LTS 40  // LDS row stride (bf16 elems): 80B, 16B-aligned, bank-spread

__global__ __launch_bounds__(256) void qkv_rope_kernel(
    const bf16* __restrict__ xb, const bf16* __restrict__ wT,
    bf16* __restrict__ qb, bf16* __restrict__ kb, bf16* __restrict__ vbT,
    float* __restrict__ kout, float* __restrict__ vout) {
  __shared__ bf16 lA[128 * LTS];
  __shared__ bf16 lB[128 * LTS];
  const int tid = threadIdx.x;
  const int wave = tid >> 5, lane = tid & 31;
  const int l16 = lane & 15, lh = lane >> 4;
  const int m0 = blockIdx.x * 128;   // row tile
  const int n0 = blockIdx.y * 128;   // col tile (== one head of q/k/v)

  v8f acc[8];
#pragma unroll
  for (int j = 0; j < 8; ++j) acc[j] = vzero8();

  for (int k0 = 0; k0 < Cc; k0 += 32) {
#pragma unroll
    for (int s = 0; s < 2; ++s) {
      int idx = tid + s * 256;               // 0..511
      int row = idx >> 2, part = idx & 3;    // 8 bf16 per part
      *(v8bf*)&lA[row * LTS + part * 8] =
          *(const v8bf*)&xb[(size_t)(m0 + row) * Cc + k0 + part * 8];
      *(v8bf*)&lB[row * LTS + part * 8] =
          *(const v8bf*)&wT[(size_t)(n0 + row) * Cc + k0 + part * 8];
    }
    __syncthreads();
    const int ra = (wave * 16 + l16) * LTS + lh * 16;
    v16bf a = ld_frag(&lA[ra], &lA[ra + 8]);             // A layout
#pragma unroll
    for (int j = 0; j < 8; ++j) {
      const int rb = (j * 16 + l16) * LTS + lh * 16;     // B layout (contig K)
      v16bf b = ld_frag(&lB[rb], &lB[rb + 8]);
      acc[j] = wmma_bf16(a, b, acc[j]);
    }
    __syncthreads();
  }

  // epilogue: this 128-wide tile is exactly one head of q, k or v
  const int type = n0 >> 11;           // 0=q 1=k 2=v
  const int h = (n0 & 2047) >> 7;
  const int mrow = m0 + wave * 16 + lh * 8;        // + i
  const int bidx = mrow >> 11, tbase = mrow & 2047;
  const int bh = bidx * Hh + h;

  if (type != 2) {  // RoPE on q/k: pair (d, d+64) == acc[j], acc[j+4]
#pragma unroll
    for (int j = 0; j < 4; ++j) {
      int d = j * 16 + l16;   // 0..63 (half-dim index)
      float invf = __expf(-(float)d * (9.210340371976184f / 64.0f));
#pragma unroll
      for (int i = 0; i < 8; ++i) {
        float s, c;
        __sincosf((float)(tbase + i) * invf, &s, &c);
        float x1 = acc[j][i], x2 = acc[j + 4][i];
        acc[j][i]     = x1 * c - x2 * s;
        acc[j + 4][i] = x1 * s + x2 * c;
      }
    }
  }

  if (type == 0) {
    const float qs = 0.08838834764831845f;  // 1/sqrt(128) folded into q
#pragma unroll
    for (int j = 0; j < 8; ++j) {
      int d = j * 16 + l16;
#pragma unroll
      for (int i = 0; i < 8; ++i)
        qb[(size_t)(bh * Tt + tbase + i) * Dd + d] = (bf16)(acc[j][i] * qs);
    }
  } else if (type == 1) {
#pragma unroll
    for (int j = 0; j < 8; ++j) {
      int d = j * 16 + l16;
#pragma unroll
      for (int i = 0; i < 8; ++i) {
        size_t idx = (size_t)(bh * Tt + tbase + i) * Dd + d;
        kout[idx] = acc[j][i];
        kb[idx] = (bf16)acc[j][i];
      }
    }
  } else {
#pragma unroll
    for (int j = 0; j < 8; ++j) {
      int d = j * 16 + l16;
      v8bf pk;
#pragma unroll
      for (int i = 0; i < 8; ++i) {
        vout[(size_t)(bh * Tt + tbase + i) * Dd + d] = acc[j][i];
        pk[i] = (bf16)acc[j][i];
      }
      // V stored [b,h,d,t] so PV B-fragments are contiguous; 16B packed store
      *(v8bf*)&vbT[((size_t)bh * Dd + d) * Tt + tbase] = pk;
    }
  }
}

// ---------------------------------------------------------------- flash attention
#define KVS 136  // TDM pad gives 272B stride: 16B-aligned, bank-spread

__global__ __launch_bounds__(256) void attn_kernel(
    const bf16* __restrict__ qb, const bf16* __restrict__ kb,
    const bf16* __restrict__ vbT, bf16* __restrict__ ob) {
  __shared__ __align__(16) bf16 lK[2][128 * KVS];
  __shared__ __align__(16) bf16 lV[2][128 * KVS];
  __shared__ __align__(16) bf16 lP[8 * 16 * KVS];
  const int qt = blockIdx.x, h = blockIdx.y, b = blockIdx.z;
  const int bh = b * Hh + h;
  const int tid = threadIdx.x;
  const int wave = tid >> 5, lane = tid & 31;
  const int l16 = lane & 15, lh = lane >> 4;

  // lK/lV are written only by the TDM; escape their addresses so the
  // optimizer cannot treat them as never-stored.
  {
    const void* e0 = (const void*)&lK[0][0];
    const void* e1 = (const void*)&lV[0][0];
    asm volatile("" ::"v"(e0), "v"(e1) : "memory");
  }

  const bf16* qbase = qb + (size_t)bh * Tt * Dd;
  const bf16* kbase = kb + (size_t)bh * Tt * Dd;
  const bf16* vbase = vbT + (size_t)bh * Dd * Tt;

  // Q fragments held in registers for the whole kv loop (A layout)
  v16bf qf[4];
  const int qrow = qt * 128 + wave * 16 + l16;
#pragma unroll
  for (int kk = 0; kk < 4; ++kk) {
    const bf16* p = qbase + (size_t)qrow * Dd + kk * 32 + lh * 16;
    qf[kk] = ld_frag(p, p + 8);
  }

  // prologue: TDM-stream tile 0 into buffer 0 (one wave issues, EXEC ignored)
  if (wave == 0) {
    tdm_load_tile_bf16((unsigned)(size_t)(void*)&lK[0][0], kbase, Dd, 128, Dd);
    tdm_load_tile_bf16((unsigned)(size_t)(void*)&lV[0][0], vbase, 128, Dd, Tt);
  }

  float m_run[8], l_run[8];
  v8f oacc[8];
#pragma unroll
  for (int i = 0; i < 8; ++i) { m_run[i] = -3.0e38f; l_run[i] = 0.0f; }
#pragma unroll
  for (int j = 0; j < 8; ++j) oacc[j] = vzero8();
  const int rbase = qt * 128 + wave * 16 + lh * 8;  // stat row = rbase + i
  bf16* lPw = &lP[wave * 16 * KVS];

  for (int kvt = 0; kvt <= qt; ++kvt) {
    const int pb = kvt & 1;
    if (wave == 0) {
      if (kvt < qt) {  // stream next tile into other buffer, then wait on ours
        tdm_load_tile_bf16((unsigned)(size_t)(void*)&lK[pb ^ 1][0],
                           kbase + (size_t)(kvt + 1) * 128 * Dd, Dd, 128, Dd);
        tdm_load_tile_bf16((unsigned)(size_t)(void*)&lV[pb ^ 1][0],
                           vbase + (kvt + 1) * 128, 128, Dd, Tt);
        __builtin_amdgcn_s_wait_tensorcnt(2);
      } else {
        __builtin_amdgcn_s_wait_tensorcnt(0);
      }
    }
    __syncthreads();
    const bf16* Kp = &lK[pb][0];
    const bf16* Vp = &lV[pb][0];

    // S = Q K^T for this 128x128 tile (wave owns 16 q-rows)
    v8f sacc[8];
#pragma unroll
    for (int j = 0; j < 8; ++j) sacc[j] = vzero8();
#pragma unroll
    for (int kk = 0; kk < 4; ++kk) {
#pragma unroll
      for (int j = 0; j < 8; ++j) {
        const int rb = (j * 16 + l16) * KVS + kk * 32 + lh * 16;
        v16bf bk = ld_frag(&Kp[rb], &Kp[rb + 8]);
        sacc[j] = wmma_bf16(qf[kk], bk, sacc[j]);
      }
    }

    if (kvt == qt) {  // causal mask on diagonal tile
#pragma unroll
      for (int j = 0; j < 8; ++j) {
        int col = kvt * 128 + j * 16 + l16;
#pragma unroll
        for (int i = 0; i < 8; ++i)
          if (col > rbase + i) sacc[j][i] = -3.0e38f;
      }
    }

    // online softmax (row spread over 16 lanes of same half + 8 accs)
#pragma unroll
    for (int i = 0; i < 8; ++i) {
      float mx = sacc[0][i];
#pragma unroll
      for (int j = 1; j < 8; ++j) mx = fmaxf(mx, sacc[j][i]);
#pragma unroll
      for (int off = 1; off < 16; off <<= 1) mx = fmaxf(mx, __shfl_xor(mx, off));
      float mnew = fmaxf(m_run[i], mx);
      float alpha = __expf(m_run[i] - mnew);
      float ls = 0.0f;
#pragma unroll
      for (int j = 0; j < 8; ++j) {
        float p = __expf(sacc[j][i] - mnew);
        sacc[j][i] = p;
        ls += p;
      }
#pragma unroll
      for (int off = 1; off < 16; off <<= 1) ls += __shfl_xor(ls, off);
      l_run[i] = l_run[i] * alpha + ls;
      m_run[i] = mnew;
#pragma unroll
      for (int j = 0; j < 8; ++j) oacc[j][i] *= alpha;
    }

    // C-layout P -> LDS -> A-layout fragments (intra-wave, no barrier needed)
#pragma unroll
    for (int i = 0; i < 8; ++i) {
      int r = lh * 8 + i;
#pragma unroll
      for (int j = 0; j < 8; ++j)
        lPw[r * KVS + j * 16 + l16] = (bf16)sacc[j][i];
    }

    // O += P V
#pragma unroll
    for (int kk = 0; kk < 4; ++kk) {
      const int ra = l16 * KVS + kk * 32 + lh * 16;
      v16bf ap = ld_frag(&lPw[ra], &lPw[ra + 8]);
#pragma unroll
      for (int j = 0; j < 8; ++j) {
        const int rb = (j * 16 + l16) * KVS + kk * 32 + lh * 16;
        v16bf bv = ld_frag(&Vp[rb], &Vp[rb + 8]);
        oacc[j] = wmma_bf16(ap, bv, oacc[j]);
      }
    }
    __syncthreads();  // all waves done reading buffer pb before it is reused
  }

  // normalize and emit attention output (bf16, [B*T, C] with head offset)
#pragma unroll
  for (int i = 0; i < 8; ++i) {
    float inv = 1.0f / l_run[i];
    int row = b * Tt + qt * 128 + wave * 16 + lh * 8 + i;
#pragma unroll
    for (int j = 0; j < 8; ++j)
      ob[(size_t)row * Cc + h * Dd + j * 16 + l16] = (bf16)(oacc[j][i] * inv);
  }
}

// ---------------------------------------------------------------- output proj
__global__ __launch_bounds__(256) void proj_kernel(
    const bf16* __restrict__ ob, const bf16* __restrict__ wT,
    float* __restrict__ out) {
  __shared__ bf16 lA[128 * LTS];
  __shared__ bf16 lB[128 * LTS];
  const int tid = threadIdx.x;
  const int wave = tid >> 5, lane = tid & 31;
  const int l16 = lane & 15, lh = lane >> 4;
  const int m0 = blockIdx.x * 128;
  const int n0 = blockIdx.y * 128;

  v8f acc[8];
#pragma unroll
  for (int j = 0; j < 8; ++j) acc[j] = vzero8();

  for (int k0 = 0; k0 < Cc; k0 += 32) {
#pragma unroll
    for (int s = 0; s < 2; ++s) {
      int idx = tid + s * 256;
      int row = idx >> 2, part = idx & 3;
      *(v8bf*)&lA[row * LTS + part * 8] =
          *(const v8bf*)&ob[(size_t)(m0 + row) * Cc + k0 + part * 8];
      *(v8bf*)&lB[row * LTS + part * 8] =
          *(const v8bf*)&wT[(size_t)(n0 + row) * Cc + k0 + part * 8];
    }
    __syncthreads();
    const int ra = (wave * 16 + l16) * LTS + lh * 16;
    v16bf a = ld_frag(&lA[ra], &lA[ra + 8]);
#pragma unroll
    for (int j = 0; j < 8; ++j) {
      const int rb = (j * 16 + l16) * LTS + lh * 16;
      v16bf b = ld_frag(&lB[rb], &lB[rb + 8]);
      acc[j] = wmma_bf16(a, b, acc[j]);
    }
    __syncthreads();
  }
#pragma unroll
  for (int j = 0; j < 8; ++j) {
#pragma unroll
    for (int i = 0; i < 8; ++i)
      out[(size_t)(m0 + wave * 16 + lh * 8 + i) * Cc + n0 + j * 16 + l16] =
          acc[j][i];
  }
}

// ---------------------------------------------------------------- launcher
extern "C" void kernel_launch(void* const* d_in, const int* in_sizes, int n_in,
                              void* d_out, int out_size, void* d_ws,
                              size_t ws_size, hipStream_t stream) {
  (void)in_sizes; (void)n_in; (void)out_size; (void)ws_size;
  const float* x      = (const float*)d_in[0];
  const float* w_qkv  = (const float*)d_in[1];
  const float* w_proj = (const float*)d_in[2];

  float* out  = (float*)d_out;                 // [B,T,C]
  float* kout = out + (size_t)Mm * Cc;         // [B,H,T,D]
  float* vout = kout + (size_t)Mm * Cc;        // [B,H,T,D]

  bf16* xb     = (bf16*)d_ws;                  // [M][C]
  bf16* wqkvT  = xb + (size_t)Mm * Cc;         // [3C][C]
  bf16* wprojT = wqkvT + (size_t)Nqkv * Cc;    // [C][C]
  bf16* qb     = wprojT + (size_t)Cc * Cc;     // [B,H,T,D] (pre-scaled, roped)
  bf16* kb     = qb + (size_t)Mm * Cc;         // [B,H,T,D] (roped)
  bf16* vbT    = kb + (size_t)Mm * Cc;         // [B,H,D,T]
  bf16* ob     = vbT + (size_t)Mm * Cc;        // [M][C] attention output

  cast_bf16_kernel<<<4096, 256, 0, stream>>>(x, xb, Mm * Cc);
  transpose_cast_kernel<<<dim3(Nqkv / 32, Cc / 32), dim3(32, 8), 0, stream>>>(
      w_qkv, wqkvT, Cc, Nqkv);
  transpose_cast_kernel<<<dim3(Cc / 32, Cc / 32), dim3(32, 8), 0, stream>>>(
      w_proj, wprojT, Cc, Cc);
  qkv_rope_kernel<<<dim3(Mm / 128, Nqkv / 128), 256, 0, stream>>>(
      xb, wqkvT, qb, kb, vbT, kout, vout);
  attn_kernel<<<dim3(Tt / 128, Hh, Bb), 256, 0, stream>>>(qb, kb, vbT, ob);
  proj_kernel<<<dim3(Mm / 128, Cc / 128), 256, 0, stream>>>(ob, wprojT, out);
}
